// GroupQueryAttention_32624571580466
// MI455X (gfx1250) — compile-verified, matrix operand order checked by reading, not run on hardware
//
#include <hip/hip_runtime.h>
#include <hip/hip_bf16.h>

typedef __attribute__((ext_vector_type(16))) __bf16 bf16x16;
typedef __attribute__((ext_vector_type(8)))  __bf16 bf16x8;
typedef __attribute__((ext_vector_type(4)))  __bf16 bf16x4;
typedef __attribute__((ext_vector_type(8)))  float  f32x8;

#define HEAD_DIM   128
#define NUM_HEADS  16
#define SEQ_T      2048
#define SEQ_K      2048
#define BATCH      2
#define BLOCK_N    64
#define NTILES     (SEQ_K / BLOCK_N)
#define WAVES_PB   8
#define BLOCK_M    (16 * WAVES_PB)
#define MODEL_DIM  (NUM_HEADS * HEAD_DIM)

__device__ __forceinline__ float redMax16(float x) {
  x = fmaxf(x, __shfl_xor(x, 1, 16));
  x = fmaxf(x, __shfl_xor(x, 2, 16));
  x = fmaxf(x, __shfl_xor(x, 4, 16));
  x = fmaxf(x, __shfl_xor(x, 8, 16));
  return x;
}
__device__ __forceinline__ float redSum16(float x) {
  x += __shfl_xor(x, 1, 16);
  x += __shfl_xor(x, 2, 16);
  x += __shfl_xor(x, 4, 16);
  x += __shfl_xor(x, 8, 16);
  return x;
}
__device__ __forceinline__ bf16x16 cat8(bf16x8 lo, bf16x8 hi8) {
  bf16x16 r;
#pragma unroll
  for (int i = 0; i < 8; ++i) { r[i] = lo[i]; r[i + 8] = hi8[i]; }
  return r;
}

__global__ __launch_bounds__(256, 1)
void gqa_flash_fwd(const float* __restrict__ Q, const float* __restrict__ Kg,
                   const float* __restrict__ Vg, float* __restrict__ Out)
{
  // Double-buffered bf16 staging: K as [k][d], V transposed as [d][k]; per-wave P slab.
  __shared__ __align__(16) __bf16 Klds[2][BLOCK_N][HEAD_DIM];     // 32 KB
  __shared__ __align__(16) __bf16 Vt[2][HEAD_DIM][BLOCK_N];       // 32 KB
  __shared__ __align__(16) __bf16 Plds[WAVES_PB][16][BLOCK_N];    // 16 KB

  const int tid  = threadIdx.x;
  const int wave = tid >> 5;
  const int lane = tid & 31;
  const int ln   = lane & 15;
  const int hi   = (lane >> 4) & 1;   // WMMA lane-half: upper half holds K 8..15 / 24..31

  const int mblocks = SEQ_T / BLOCK_M;
  const int mblk = blockIdx.x % mblocks;
  const int h    = (blockIdx.x / mblocks) % NUM_HEADS;
  const int b    = blockIdx.x / (mblocks * NUM_HEADS);

  const int qrow = mblk * BLOCK_M + wave * 16 + ln;
  const float scale = 0.08838834764831845f;  // 1/sqrt(128)

  // ---- Q fragments (A-layout, 16x32 bf16 per 32-deep k-chunk), pre-scaled, resident
  bf16x16 qa[4];
  {
    const float* qp = Q + (size_t)(b * SEQ_T + qrow) * MODEL_DIM + h * HEAD_DIM;
#pragma unroll
    for (int c = 0; c < 4; ++c) {
      const int base = c * 32 + hi * 8;
      float4 x0 = *(const float4*)(qp + base);
      float4 x1 = *(const float4*)(qp + base + 4);
      float4 y0 = *(const float4*)(qp + base + 16);
      float4 y1 = *(const float4*)(qp + base + 20);
      float t[16] = {x0.x,x0.y,x0.z,x0.w, x1.x,x1.y,x1.z,x1.w,
                     y0.x,y0.y,y0.z,y0.w, y1.x,y1.y,y1.z,y1.w};
      bf16x16 a;
#pragma unroll
      for (int i = 0; i < 16; ++i) a[i] = (__bf16)(t[i] * scale);
      qa[c] = a;
    }
  }

  // ---- O accumulator (16x128 f32 -> 8 C-tiles) + online-softmax state
  f32x8 Oacc[8];
#pragma unroll
  for (int dt = 0; dt < 8; ++dt) { f32x8 z = {}; Oacc[dt] = z; }
  float m_s[8], l_s[8], alpha[8];
#pragma unroll
  for (int r = 0; r < 8; ++r) { m_s[r] = -1.0e30f; l_s[r] = 0.0f; }

  // Cooperative staging map: 64 rows x 128 cols; 4 threads per row, 32 f32 each.
  const int srow = tid >> 2;            // 0..63
  const int scol = (tid & 3) * 32;      // 0,32,64,96
  const size_t kv_base = (size_t)b * SEQ_K * MODEL_DIM + (size_t)h * HEAD_DIM;

  auto stage = [&](int tile, int buf) {
    const int kglob = tile * BLOCK_N + srow;
    const float* kp = Kg + kv_base + (size_t)kglob * MODEL_DIM + scol;
    const float* vp = Vg + kv_base + (size_t)kglob * MODEL_DIM + scol;
#pragma unroll
    for (int v4 = 0; v4 < 8; ++v4) {
      float4 kv = *(const float4*)(kp + v4 * 4);
      float4 vv = *(const float4*)(vp + v4 * 4);
      const int c0 = scol + v4 * 4;
      bf16x4 kb4 = { (__bf16)kv.x, (__bf16)kv.y, (__bf16)kv.z, (__bf16)kv.w };
      *(bf16x4*)&Klds[buf][srow][c0] = kb4;           // ds_store_b64
      Vt[buf][c0 + 0][srow] = (__bf16)vv.x;           // transpose scatter
      Vt[buf][c0 + 1][srow] = (__bf16)vv.y;
      Vt[buf][c0 + 2][srow] = (__bf16)vv.z;
      Vt[buf][c0 + 3][srow] = (__bf16)vv.w;
    }
    if (tile + 1 < NTILES) {
      __builtin_prefetch(kp + (size_t)BLOCK_N * MODEL_DIM, 0, 0);
      __builtin_prefetch(vp + (size_t)BLOCK_N * MODEL_DIM, 0, 0);
    }
  };

  // Prologue: stage tile 0 into buffer 0.
  stage(0, 0);
  __syncthreads();

  for (int kt = 0; kt < NTILES; ++kt) {
    const int cur = kt & 1;
    // Overlap: issue next tile's global loads + LDS stores before consuming current tile.
    if (kt + 1 < NTILES) stage(kt + 1, cur ^ 1);

    // ---- S = Q * K^T : four 16x16 subtiles, K-depth 128 chained over 4 WMMAs each
    f32x8 s[4];
#pragma unroll
    for (int j = 0; j < 4; ++j) {
      f32x8 acc = {};
#pragma unroll
      for (int c = 0; c < 4; ++c) {
        const int kb = c * 32 + hi * 8;
        bf16x16 bk = cat8(*(const bf16x8*)&Klds[cur][j * 16 + ln][kb],
                          *(const bf16x8*)&Klds[cur][j * 16 + ln][kb + 16]);
        acc = __builtin_amdgcn_wmma_f32_16x16x32_bf16(false, qa[c], false, bk,
                                                      (short)0, acc, false, false);
      }
      s[j] = acc;
    }

    // ---- online softmax (fp32); reductions amortized over 64 keys
#pragma unroll
    for (int r = 0; r < 8; ++r) {
      float cur4 = fmaxf(fmaxf(s[0][r], s[1][r]), fmaxf(s[2][r], s[3][r]));
      float mnew = fmaxf(m_s[r], redMax16(cur4));
      float a    = __expf(m_s[r] - mnew);
      float p0   = __expf(s[0][r] - mnew);
      float p1   = __expf(s[1][r] - mnew);
      float p2   = __expf(s[2][r] - mnew);
      float p3   = __expf(s[3][r] - mnew);
      l_s[r]   = l_s[r] * a + redSum16((p0 + p1) + (p2 + p3));
      m_s[r]   = mnew;
      alpha[r] = a;
      const int mr = r + hi * 8;
      Plds[wave][mr][ln]      = (__bf16)p0;   // C-layout -> LDS [m][n]
      Plds[wave][mr][ln + 16] = (__bf16)p1;
      Plds[wave][mr][ln + 32] = (__bf16)p2;
      Plds[wave][mr][ln + 48] = (__bf16)p3;
    }
#pragma unroll
    for (int dt = 0; dt < 8; ++dt)
#pragma unroll
      for (int r = 0; r < 8; ++r) Oacc[dt][r] *= alpha[r];

    // ---- P back out of LDS in A-layout (two 32-deep k-chunks); O += P * V
    const int pb = hi * 8;
    bf16x16 pa[2];
#pragma unroll
    for (int jj = 0; jj < 2; ++jj)
      pa[jj] = cat8(*(const bf16x8*)&Plds[wave][ln][jj * 32 + pb],
                    *(const bf16x8*)&Plds[wave][ln][jj * 32 + pb + 16]);
#pragma unroll
    for (int dt = 0; dt < 8; ++dt) {
#pragma unroll
      for (int jj = 0; jj < 2; ++jj) {
        bf16x16 vb = cat8(*(const bf16x8*)&Vt[cur][dt * 16 + ln][jj * 32 + pb],
                          *(const bf16x8*)&Vt[cur][dt * 16 + ln][jj * 32 + pb + 16]);
        Oacc[dt] = __builtin_amdgcn_wmma_f32_16x16x32_bf16(false, pa[jj], false, vb,
                                                           (short)0, Oacc[dt], false, false);
      }
    }

    // Single barrier per iteration: next-tile staging visible AND current buffer free.
    __syncthreads();
  }

  // ---- epilogue: O / l, fp32 stores
#pragma unroll
  for (int r = 0; r < 8; ++r) {
    const float inv = 1.0f / l_s[r];
    const int mr = r + hi * 8;
    float* op = Out + (size_t)(b * SEQ_T + mblk * BLOCK_M + wave * 16 + mr) * MODEL_DIM
                    + h * HEAD_DIM + ln;
#pragma unroll
    for (int dt = 0; dt < 8; ++dt) op[dt * 16] = Oacc[dt][r] * inv;
  }
}

extern "C" void kernel_launch(void* const* d_in, const int* in_sizes, int n_in,
                              void* d_out, int out_size, void* d_ws, size_t ws_size,
                              hipStream_t stream) {
  (void)in_sizes; (void)n_in; (void)out_size; (void)d_ws; (void)ws_size;
  const float* Q = (const float*)d_in[0];
  const float* K = (const float*)d_in[1];
  const float* V = (const float*)d_in[2];
  float* Out = (float*)d_out;
  dim3 grid(BATCH * NUM_HEADS * (SEQ_T / BLOCK_M));  // 2*16*16 = 512 workgroups
  gqa_flash_fwd<<<grid, 256, 0, stream>>>(Q, K, V, Out);
}
